// FullyAttention_34918084116494
// MI455X (gfx1250) — matrix operand order repaired and checked by reading, not compiled, not run on hardware
//
#include <hip/hip_runtime.h>

// ---- problem constants (from reference) ----
#define B_  8
#define L_  128
#define Q_  8
#define H_  8
#define E_  64
#define S_  128
#define P_  8
#define M_  (L_*Q_)        // 1024 query rows per head
#define N_  (S_*P_)        // 1024 key rows per head
#define ROWSTRIDE (H_*E_)  // 512 floats between consecutive m/n rows
#define HEADS (B_*H_)      // 64

#define NT    64           // keys processed per N-step
#define NSTEPS (N_/NT)     // 16
#define WG_M  128          // query rows per workgroup (8 waves x 16)
#define WAVES 8
#define SKS   72           // LDS row stride, K tile (bf16 elems, padded)
#define SVS   72           // LDS row stride, transposed V tile
#define SPS   72           // LDS row stride, per-wave P tile

typedef __attribute__((ext_vector_type(16))) __bf16 v16bf;
typedef __attribute__((ext_vector_type(8)))  float  v8f;

union F16 {
    v16bf v;
    unsigned short s[16];
    unsigned long long q[4];
};

__device__ __forceinline__ unsigned short bfbits(float x) {
    __bf16 h = (__bf16)x;           // native f32->bf16 (RNE)
    unsigned short u;
    __builtin_memcpy(&u, &h, 2);
    return u;
}

// 16-byte async copy global -> LDS (tracked by ASYNCcnt)
__device__ __forceinline__ void async_cp16(unsigned lds_off, const void* gptr) {
    asm volatile("global_load_async_to_lds_b128 %0, %1, off"
                 :: "v"(lds_off), "v"(gptr) : "memory");
}

// max-reduction across xor partners within each 16-lane row (VALU permlane path)
__device__ __forceinline__ float redmax16(float x) {
#if __has_builtin(__builtin_amdgcn_permlane16)
    unsigned u;
    u = __builtin_amdgcn_permlane16(__float_as_uint(x), __float_as_uint(x),
                                    0x67452301u, 0xEFCDAB89u, false, false);   // xor 1
    x = fmaxf(x, __uint_as_float(u));
    u = __builtin_amdgcn_permlane16(__float_as_uint(x), __float_as_uint(x),
                                    0x54761032u, 0xDCFE98BAu, false, false);   // xor 2
    x = fmaxf(x, __uint_as_float(u));
    u = __builtin_amdgcn_permlane16(__float_as_uint(x), __float_as_uint(x),
                                    0x32107654u, 0xBA98FEDCu, false, false);   // xor 4
    x = fmaxf(x, __uint_as_float(u));
    u = __builtin_amdgcn_permlane16(__float_as_uint(x), __float_as_uint(x),
                                    0xFEDCBA98u, 0x76543210u, false, false);   // xor 8
    x = fmaxf(x, __uint_as_float(u));
#else
    x = fmaxf(x, __shfl_xor(x, 1, 32));
    x = fmaxf(x, __shfl_xor(x, 2, 32));
    x = fmaxf(x, __shfl_xor(x, 4, 32));
    x = fmaxf(x, __shfl_xor(x, 8, 32));
#endif
    return x;
}

// ============================================================================
// Kernel 1: one-time layout/precision conversion
//   K  -> bf16 [head][n][e]   ;   V -> bf16 [head][e][n] (transposed)
// ============================================================================
__global__ __launch_bounds__(256) void cvt_kv(
    const float* __restrict__ kg, const float* __restrict__ vg,
    unsigned short* __restrict__ kbf, unsigned short* __restrict__ vtbf)
{
    const int g   = blockIdx.x * 256 + threadIdx.x;  // 0 .. HEADS*N_*8-1
    const int seg = (g & 7) * 8;                     // e segment start
    const int n   = (g >> 3) & (N_ - 1);
    const int head = g >> 13;
    const int b = head >> 3, h = head & 7;
    const size_t src = (((size_t)b * N_ + n) * H_ + h) * E_ + seg;

    unsigned short* kd = kbf + ((size_t)head * N_ + n) * E_ + seg;
    #pragma unroll
    for (int j = 0; j < 8; ++j) kd[j] = bfbits(kg[src + j]);
    #pragma unroll
    for (int j = 0; j < 8; ++j)
        vtbf[((size_t)head * E_ + seg + j) * N_ + n] = bfbits(vg[src + j]);
}

// ============================================================================
// Kernel 2: fused flash attention, bf16 WMMA, async double-buffered tiles
// ============================================================================
__global__ __launch_bounds__(256) void fused_attn_bf16_wmma(
    const float* __restrict__ qg,
    const unsigned short* __restrict__ kbf,
    const unsigned short* __restrict__ vtbf,
    float* __restrict__ og)
{
    __shared__ unsigned short sKt[2][NT * SKS];      // 64 x 64 bf16, padded rows
    __shared__ unsigned short sVt[2][E_ * SVS];      // 64 x 64 bf16 (e-major)
    __shared__ unsigned short sP[WAVES][16 * SPS];   // per-wave 16 x 64 bf16 P

    const int mtile = blockIdx.x;      // 0..7
    const int head  = blockIdx.y;      // 0..63
    const int b = head >> 3, h = head & 7;

    const int tid   = threadIdx.x;
    const int wave  = tid >> 5;
    const int lane  = tid & 31;
    const int lhalf = lane >> 4;
    const int l15   = lane & 15;

    const float* Qb = qg + ((size_t)b * M_ * H_ + h) * E_;
    float*       Ob = og + ((size_t)b * M_ * H_ + h) * E_;
    const unsigned short* Kh  = kbf  + (size_t)head * N_ * E_;
    const unsigned short* Vth = vtbf + (size_t)head * E_ * N_;

    // -------- load Q A-fragments (wave's 16 rows, all 64 e) --------
    const int m0 = mtile * WG_M + wave * 16;
    const float qscale = 0.125f * 1.44269504088896f;   // 1/sqrt(E) * log2(e)
    F16 aQ0, aQ1;
    {
        const float* rp = Qb + (size_t)(m0 + l15) * ROWSTRIDE;
        const int kb = lhalf * 16;
        #pragma unroll
        for (int j = 0; j < 8; ++j) {
            aQ0.s[j]     = bfbits(qscale * rp[kb + j]);
            aQ0.s[8 + j] = bfbits(qscale * rp[8 + kb + j]);
            aQ1.s[j]     = bfbits(qscale * rp[32 + kb + j]);
            aQ1.s[8 + j] = bfbits(qscale * rp[40 + kb + j]);
        }
    }

    // all-ones B fragment (for row sums of P via WMMA)
    F16 bOnes;
    #pragma unroll
    for (int i = 0; i < 16; ++i) bOnes.s[i] = 0x3F80;   // bf16(1.0)

    v8f acc[4] = {};
    float mrun[8], lrun[8];
    #pragma unroll
    for (int r = 0; r < 8; ++r) { mrun[r] = -1.0e30f; lrun[r] = 0.0f; }
    const v8f zacc = {};

    // -------- async staging: 2 chunks each for K and Vt per thread --------
    // tile = 64 rows x 128 bytes = 512 x 16B chunks; 256 threads -> 2 chunks each
    const int c0row = tid >> 3,          c0ch = tid & 7;
    const int c1row = (tid + 256) >> 3,  c1ch = tid & 7;

    #define STAGE_TILE(nb_, buf_)                                                          \
        do {                                                                               \
            async_cp16((unsigned)(uintptr_t)((const char*)sKt[buf_] + c0row*(SKS*2) + c0ch*16), \
                       (const char*)(Kh + (size_t)((nb_) + c0row) * E_) + c0ch*16);        \
            async_cp16((unsigned)(uintptr_t)((const char*)sKt[buf_] + c1row*(SKS*2) + c1ch*16), \
                       (const char*)(Kh + (size_t)((nb_) + c1row) * E_) + c1ch*16);        \
            async_cp16((unsigned)(uintptr_t)((const char*)sVt[buf_] + c0row*(SVS*2) + c0ch*16), \
                       (const char*)(Vth + (size_t)c0row * N_ + (nb_)) + c0ch*16);         \
            async_cp16((unsigned)(uintptr_t)((const char*)sVt[buf_] + c1row*(SVS*2) + c1ch*16), \
                       (const char*)(Vth + (size_t)c1row * N_ + (nb_)) + c1ch*16);         \
        } while (0)

    STAGE_TILE(0, 0);

    for (int step = 0; step < NSTEPS; ++step) {
        const int nb  = step * NT;
        const int cur = step & 1;

        if (step + 1 < NSTEPS) {
            STAGE_TILE(nb + NT, cur ^ 1);
            asm volatile("s_wait_asynccnt 4" ::: "memory");   // current tile landed
        } else {
            asm volatile("s_wait_asynccnt 0" ::: "memory");
        }
        __syncthreads();

        const unsigned short* sk = sKt[cur];
        const unsigned short* sv = sVt[cur];
        const int kb2 = lhalf * 16;

        // ---------- S = Q * K^T : 4 col-tiles of 16 keys, K-dim 64 ----------
        v8f s[4];
        #pragma unroll
        for (int ct = 0; ct < 4; ++ct) {
            F16 bk;
            {
                const unsigned short* p = &sk[(ct * 16 + l15) * SKS + kb2];
                bk.q[0] = *(const unsigned long long*)(p);
                bk.q[1] = *(const unsigned long long*)(p + 4);
                bk.q[2] = *(const unsigned long long*)(p + 8);
                bk.q[3] = *(const unsigned long long*)(p + 12);
            }
            s[ct] = __builtin_amdgcn_wmma_f32_16x16x32_bf16(false, aQ0.v, false, bk.v,
                                                            (short)0, zacc, false, false);
            {
                const unsigned short* p = &sk[(ct * 16 + l15) * SKS + 32 + kb2];
                bk.q[0] = *(const unsigned long long*)(p);
                bk.q[1] = *(const unsigned long long*)(p + 4);
                bk.q[2] = *(const unsigned long long*)(p + 8);
                bk.q[3] = *(const unsigned long long*)(p + 12);
            }
            s[ct] = __builtin_amdgcn_wmma_f32_16x16x32_bf16(false, aQ1.v, false, bk.v,
                                                            (short)0, s[ct], false, false);
        }

        // ---------- online softmax: max via permlane, P -> LDS ----------
        float alpha[8];
        unsigned short* sp = sP[wave];
        #pragma unroll
        for (int r = 0; r < 8; ++r) {
            float x = fmaxf(fmaxf(s[0][r], s[1][r]), fmaxf(s[2][r], s[3][r]));
            x = redmax16(x);
            const float mnew = fmaxf(mrun[r], x);
            alpha[r] = __builtin_amdgcn_exp2f(mrun[r] - mnew);
            mrun[r] = mnew;
            const int rr = r + lhalf * 8;
            #pragma unroll
            for (int ct = 0; ct < 4; ++ct) {
                const float p = __builtin_amdgcn_exp2f(s[ct][r] - mnew);
                sp[rr * SPS + ct * 16 + l15] = bfbits(p);
            }
        }

        // ---------- reload P in A-layout (two K-chunks of 32) ----------
        F16 aP0, aP1;
        {
            const int kb = lhalf * 16;
            const unsigned short* p = &sp[l15 * SPS];
            aP0.q[0] = *(const unsigned long long*)(p + kb);
            aP0.q[1] = *(const unsigned long long*)(p + kb + 4);
            aP0.q[2] = *(const unsigned long long*)(p + 8 + kb);
            aP0.q[3] = *(const unsigned long long*)(p + 8 + kb + 4);
            aP1.q[0] = *(const unsigned long long*)(p + 32 + kb);
            aP1.q[1] = *(const unsigned long long*)(p + 32 + kb + 4);
            aP1.q[2] = *(const unsigned long long*)(p + 40 + kb);
            aP1.q[3] = *(const unsigned long long*)(p + 40 + kb + 4);
        }

        // ---------- row sums of P via WMMA against all-ones B ----------
        v8f sums = __builtin_amdgcn_wmma_f32_16x16x32_bf16(false, aP0.v, false, bOnes.v,
                                                           (short)0, zacc, false, false);
        sums = __builtin_amdgcn_wmma_f32_16x16x32_bf16(false, aP1.v, false, bOnes.v,
                                                       (short)0, sums, false, false);
        #pragma unroll
        for (int r = 0; r < 8; ++r) lrun[r] = alpha[r] * lrun[r] + sums[r];

        // ---------- rescale O and accumulate O += P * V ----------
        #pragma unroll
        for (int ec = 0; ec < 4; ++ec) {
            #pragma unroll
            for (int r = 0; r < 8; ++r) acc[ec][r] *= alpha[r];
        }
        #pragma unroll
        for (int ec = 0; ec < 4; ++ec) {
            F16 bv;
            const unsigned short* p0 = &sv[(ec * 16 + l15) * SVS + kb2];
            bv.q[0] = *(const unsigned long long*)(p0);
            bv.q[1] = *(const unsigned long long*)(p0 + 4);
            bv.q[2] = *(const unsigned long long*)(p0 + 8);
            bv.q[3] = *(const unsigned long long*)(p0 + 12);
            acc[ec] = __builtin_amdgcn_wmma_f32_16x16x32_bf16(false, aP0.v, false, bv.v,
                                                              (short)0, acc[ec], false, false);
            const unsigned short* p1 = p0 + 32;
            bv.q[0] = *(const unsigned long long*)(p1);
            bv.q[1] = *(const unsigned long long*)(p1 + 4);
            bv.q[2] = *(const unsigned long long*)(p1 + 8);
            bv.q[3] = *(const unsigned long long*)(p1 + 12);
            acc[ec] = __builtin_amdgcn_wmma_f32_16x16x32_bf16(false, aP1.v, false, bv.v,
                                                              (short)0, acc[ec], false, false);
        }
        __syncthreads();   // everyone done reading cur before it is overwritten
    }

    // ---------- finalize ----------
    #pragma unroll
    for (int r = 0; r < 8; ++r) {
        const int m = m0 + r + lhalf * 8;
        const float inv = 1.0f / lrun[r];
        float* orow = Ob + (size_t)m * ROWSTRIDE;
        #pragma unroll
        for (int ec = 0; ec < 4; ++ec)
            orow[ec * 16 + l15] = acc[ec][r] * inv;
    }
}

extern "C" void kernel_launch(void* const* d_in, const int* in_sizes, int n_in,
                              void* d_out, int out_size, void* d_ws, size_t ws_size,
                              hipStream_t stream) {
    (void)in_sizes; (void)n_in; (void)out_size; (void)ws_size;
    const float* q = (const float*)d_in[0];
    const float* k = (const float*)d_in[1];
    const float* v = (const float*)d_in[2];
    float* out = (float*)d_out;

    unsigned short* kbf  = (unsigned short*)d_ws;                       // 8 MB
    unsigned short* vtbf = kbf + (size_t)HEADS * N_ * E_;               // 8 MB

    // pass 1: precision/layout conversion (K row-major bf16, V transposed bf16)
    cvt_kv<<<(HEADS * N_ * 8) / 256, 256, 0, stream>>>(k, v, kbf, vtbf);

    // pass 2: fused attention
    dim3 grid(M_ / WG_M, HEADS);   // 8 m-tiles x 64 heads
    fused_attn_bf16_wmma<<<grid, 256, 0, stream>>>(q, kbf, vtbf, out);
}
